// MSHGNBlock_55757265437218
// MI455X (gfx1250) — matrix operand order repaired and checked by reading, not compiled, not run on hardware
//
#include <hip/hip_runtime.h>
#include <cstdint>
#include <cstddef>

// ---------------------------------------------------------------------------
// MSHGN block for MI455X (gfx1250, wave32). All linear/pointwise-conv/FFN
// matmuls run through v_wmma_f32_16x16x32_f16 (f16 A/B, f32 accumulate).
// Elementwise / LN / softmax stages are f32 bandwidth kernels.
// ---------------------------------------------------------------------------

typedef __attribute__((ext_vector_type(16))) _Float16 v16h;
typedef __attribute__((ext_vector_type(8)))  float    v8f;

constexpr int cB  = 8;
constexpr int cC  = 64;
constexpr int cL  = 512;
constexpr int cD  = 128;
constexpr int cH  = 4;
constexpr int cDH = 32;      // cD / cH
constexpr int cKC = 9;       // K_CONV
constexpr int cBC = cB * cC; // 512

__device__ __forceinline__ float gelu_f(float x) {
  // exact erf GELU (torch nn.GELU default)
  return 0.5f * x * (1.0f + erff(x * 0.7071067811865475f));
}

// ---------------------------------------------------------------------------
// Weight prep: f32 [K,N] (transpose=1) or f32 [N,K] (transpose=0) -> f16 [N,K]
// ---------------------------------------------------------------------------
__global__ void k_prep_w(const float* __restrict__ src, _Float16* __restrict__ dst,
                         int K, int N, int transpose) {
  int idx = blockIdx.x * blockDim.x + threadIdx.x;
  if (idx >= K * N) return;
  int n = idx / K;
  int k = idx - n * K;
  float v = transpose ? src[(size_t)k * N + n] : src[(size_t)n * K + k];
  dst[idx] = (_Float16)v;
}

// ---------------------------------------------------------------------------
// WMMA GEMM: C[M,N] = act(A[M,K] * Bt[N,K]^T + bias)
// A: f32 row-major (converted to f16 on the fly), Bt: f16 [N,K].
// One wave computes a 16x64 strip: A-fragment reused across 4 B-fragments.
// M % 16 == 0, N % 64 == 0, K % 32 == 0 (all shapes here satisfy this).
// act: 0 = none, 1 = GELU. bias may be nullptr.
// ---------------------------------------------------------------------------
__global__ void k_wmma_gemm(const float* __restrict__ A,
                            const _Float16* __restrict__ Bt,
                            const float* __restrict__ bias,
                            float* __restrict__ Cmat,
                            int M, int N, int K, int act) {
  int gtid = blockIdx.x * blockDim.x + threadIdx.x;
  int wave = gtid >> 5;
  int lane = threadIdx.x & 31;
  int mtiles  = M >> 4;
  int ngroups = N >> 6;
  if (wave >= mtiles * ngroups) return;
  int mt    = wave / ngroups;
  int ng    = wave - mt * ngroups;
  int m0    = mt << 4;
  int nbase = ng << 6;

  int halfsel = lane >> 4;       // 0: lanes 0-15, 1: lanes 16-31
  int lmod    = lane & 15;
  int arow    = m0 + lmod;       // A: row M = lane%16
  int akoff   = halfsel << 4;    // A: K offset 0 or 16 (contiguous, matches B)
  int bkoff   = halfsel << 4;    // B: K offset 0 or 16 (contiguous, matches A)

  v8f acc[4] = {};

  const float* arowp = A + (size_t)arow * K;
  for (int k0 = 0; k0 < K; k0 += 32) {
    const float* ap = arowp + k0;
    if (k0 + 32 < K) __builtin_prefetch(ap + 32, 0, 1);  // global_prefetch_b8
    v16h afrag;
#pragma unroll
    for (int e = 0; e < 8; ++e) afrag[e] = (_Float16)ap[akoff + e];
#pragma unroll
    for (int e = 0; e < 8; ++e) afrag[8 + e] = (_Float16)ap[akoff + 8 + e];
#pragma unroll
    for (int t = 0; t < 4; ++t) {
      const _Float16* bp = Bt + (size_t)(nbase + (t << 4) + lmod) * K + k0 + bkoff;
      v16h bfrag = *(const v16h*)bp;   // 32B aligned: K%32==0, bkoff in {0,16}
      acc[t] = __builtin_amdgcn_wmma_f32_16x16x32_f16(
          false, afrag, false, bfrag, (short)0, acc[t], false, false);
    }
  }

  int row0 = m0 + (halfsel << 3);
#pragma unroll
  for (int t = 0; t < 4; ++t) {
    int col = nbase + (t << 4) + lmod;
    float bv = bias ? bias[col] : 0.0f;
#pragma unroll
    for (int r = 0; r < 8; ++r) {
      float v = acc[t][r] + bv;
      if (act) v = gelu_f(v);
      Cmat[(size_t)(row0 + r) * N + col] = v;
    }
  }
}

// ---------------------------------------------------------------------------
// Depthwise conv along L (zero 'same' padding) + GELU. Layout [Nb, Ll, D].
// w: [D, 1, ksz] flat d*ksz+j.
// ---------------------------------------------------------------------------
__global__ void k_dwconv_gelu(const float* __restrict__ X,
                              const float* __restrict__ W,
                              const float* __restrict__ Bv,
                              float* __restrict__ Y,
                              int Nb, int Ll, int ksz) {
  int idx = blockIdx.x * blockDim.x + threadIdx.x;
  int total = Nb * Ll * cD;
  if (idx >= total) return;
  int d = idx & (cD - 1);
  int t = idx >> 7;
  int l = t % Ll;
  int n = t / Ll;
  int half = ksz >> 1;
  float s = Bv[d];
  for (int j = 0; j < ksz; ++j) {
    int ll = l + j - half;
    if (ll >= 0 && ll < Ll)
      s += X[((size_t)n * Ll + ll) * cD + d] * W[d * ksz + j];
  }
  Y[idx] = gelu_f(s);
}

// ---------------------------------------------------------------------------
// Series decomposition: 9-tap moving average with replicate padding.
// ---------------------------------------------------------------------------
__global__ void k_decomp(const float* __restrict__ Hh,
                         float* __restrict__ Sea, float* __restrict__ Tre,
                         int Nb, int Ll) {
  int idx = blockIdx.x * blockDim.x + threadIdx.x;
  int total = Nb * Ll * cD;
  if (idx >= total) return;
  int d = idx & (cD - 1);
  int t = idx >> 7;
  int l = t % Ll;
  int n = t / Ll;
  float s = 0.0f;
  for (int j = -4; j <= 4; ++j) {
    int ll = l + j;
    ll = ll < 0 ? 0 : (ll >= Ll ? Ll - 1 : ll);
    s += Hh[((size_t)n * Ll + ll) * cD + d];
  }
  float tr = s * (1.0f / 9.0f);
  float hv = Hh[idx];
  Sea[idx] = hv - tr;
  Tre[idx] = tr;
}

__global__ void k_avgpool2(const float* __restrict__ X, float* __restrict__ Y,
                           int Nb, int Lin) {
  int Lout = Lin >> 1;
  int idx = blockIdx.x * blockDim.x + threadIdx.x;
  int total = Nb * Lout * cD;
  if (idx >= total) return;
  int d = idx & (cD - 1);
  int t = idx >> 7;
  int l = t % Lout;
  int n = t / Lout;
  const float* xb = X + ((size_t)n * Lin + 2 * l) * cD + d;
  Y[idx] = 0.5f * (xb[0] + xb[cD]);
}

__global__ void k_interp2x(const float* __restrict__ X, float* __restrict__ Y,
                           int Nb, int Lin) {
  int Lout = Lin << 1;
  int idx = blockIdx.x * blockDim.x + threadIdx.x;
  int total = Nb * Lout * cD;
  if (idx >= total) return;
  int d = idx & (cD - 1);
  int t = idx >> 7;
  int l = t % Lout;
  int n = t / Lout;
  float coord = 0.5f * (float)l - 0.25f;           // (l+0.5)*Lin/Lout - 0.5
  coord = fminf(fmaxf(coord, 0.0f), (float)(Lin - 1));
  int lo = (int)floorf(coord);
  int hi = lo + 1; if (hi > Lin - 1) hi = Lin - 1;
  float w = coord - (float)lo;
  const float* xb = X + (size_t)n * Lin * cD + d;
  Y[idx] = xb[(size_t)lo * cD] * (1.0f - w) + xb[(size_t)hi * cD] * w;
}

__global__ void k_add(const float* __restrict__ A, const float* __restrict__ Bp,
                      float* __restrict__ Out, int n) {
  int idx = blockIdx.x * blockDim.x + threadIdx.x;
  if (idx >= n) return;
  Out[idx] = A[idx] + Bp[idx];
}

// ---------------------------------------------------------------------------
// LayerNorm over last dim (128). One wave32 per row, 4 elems per lane,
// shuffle reductions. Optional pre-add (Out = LN(A + Add)).
// ---------------------------------------------------------------------------
__global__ void k_ln(const float* __restrict__ A, const float* __restrict__ Add,
                     const float* __restrict__ g, const float* __restrict__ be,
                     float* __restrict__ Out, int rows) {
  int wave = (blockIdx.x * blockDim.x + threadIdx.x) >> 5;
  int lane = threadIdx.x & 31;
  if (wave >= rows) return;
  size_t base = (size_t)wave * cD;
  float v4[4];
  float s = 0.0f;
#pragma unroll
  for (int i = 0; i < 4; ++i) {
    int e = lane + 32 * i;
    float t = A[base + e];
    if (Add) t += Add[base + e];
    v4[i] = t;
    s += t;
  }
#pragma unroll
  for (int off = 16; off > 0; off >>= 1) s += __shfl_xor(s, off, 32);
  float mean = s * (1.0f / 128.0f);
  float var = 0.0f;
#pragma unroll
  for (int i = 0; i < 4; ++i) {
    float dlt = v4[i] - mean;
    var += dlt * dlt;
  }
#pragma unroll
  for (int off = 16; off > 0; off >>= 1) var += __shfl_xor(var, off, 32);
  var *= (1.0f / 128.0f);
  float inv = rsqrtf(var + 1e-5f);
#pragma unroll
  for (int i = 0; i < 4; ++i) {
    int e = lane + 32 * i;
    Out[base + e] = (v4[i] - mean) * inv * g[e] + be[e];
  }
}

// ---------------------------------------------------------------------------
// Temporal attention: one thread per (b, l, h, cq). dh = 32, C = 64 keys.
// Q/K/V stored as [B,C,L,D] (same indexing as x).
// ---------------------------------------------------------------------------
__global__ void k_tattn(const float* __restrict__ Q, const float* __restrict__ Kt,
                        const float* __restrict__ V, const float* __restrict__ mask,
                        float* __restrict__ Out) {
  int idx = blockIdx.x * blockDim.x + threadIdx.x;
  if (idx >= cB * cL * cH * cC) return;
  int cq = idx & (cC - 1);
  int t  = idx >> 6;
  int h  = t & (cH - 1);
  t >>= 2;
  int l = t & (cL - 1);
  int b = t >> 9;

  float q[cDH];
  size_t qb = ((size_t)(b * cC + cq) * cL + l) * cD + h * cDH;
  for (int j = 0; j < cDH; ++j) q[j] = Q[qb + j];

  float sc[cC];
  float mx = -3.402823466e38f;
  const float scale = 0.17677669529663687f;  // 32^-0.5
  for (int ck = 0; ck < cC; ++ck) {
    size_t kb = ((size_t)(b * cC + ck) * cL + l) * cD + h * cDH;
    float s = 0.0f;
    for (int j = 0; j < cDH; ++j) s += q[j] * Kt[kb + j];
    s *= scale;
    float mv = mask[((size_t)(b * cC + ck)) * cL + l];
    if (mv == 0.0f) s = -3.402823466e38f;
    sc[ck] = s;
    mx = fmaxf(mx, s);
  }
  float sum = 0.0f;
  for (int ck = 0; ck < cC; ++ck) {
    float e = expf(sc[ck] - mx);
    sc[ck] = e;
    sum += e;
  }
  float inv = 1.0f / sum;
  float o[cDH];
  for (int j = 0; j < cDH; ++j) o[j] = 0.0f;
  for (int ck = 0; ck < cC; ++ck) {
    float w = sc[ck] * inv;
    size_t vb = ((size_t)(b * cC + ck) * cL + l) * cD + h * cDH;
    for (int j = 0; j < cDH; ++j) o[j] += w * V[vb + j];
  }
  for (int j = 0; j < cDH; ++j) Out[qb + j] = o[j];
}

__global__ void k_msum_c(const float* __restrict__ mask, float* __restrict__ msumc) {
  int idx = blockIdx.x * blockDim.x + threadIdx.x;  // B*L
  if (idx >= cB * cL) return;
  int l = idx & (cL - 1);
  int b = idx >> 9;
  float s = 0.0f;
  for (int c = 0; c < cC; ++c) s += mask[((size_t)(b * cC + c)) * cL + l];
  msumc[idx] = s;
}

__global__ void k_msum_l(const float* __restrict__ mask, float* __restrict__ msuml) {
  int idx = blockIdx.x * blockDim.x + threadIdx.x;  // B*C
  if (idx >= cB * cC) return;
  float s = 0.0f;
  const float* mp = mask + (size_t)idx * cL;
  for (int l = 0; l < cL; ++l) s += mp[l];
  msuml[idx] = s;
}

// E_t_raw[b,l,d] = sum_c X[b,c,l,d]*mask[b,c,l] / max(msumc,1)
__global__ void k_mean_c(const float* __restrict__ X, const float* __restrict__ mask,
                         const float* __restrict__ msumc, float* __restrict__ Out) {
  int idx = blockIdx.x * blockDim.x + threadIdx.x;  // B*L*D
  if (idx >= cB * cL * cD) return;
  int d = idx & (cD - 1);
  int t = idx >> 7;
  int l = t & (cL - 1);
  int b = t >> 9;
  float s = 0.0f;
  for (int c = 0; c < cC; ++c) {
    float mv = mask[((size_t)(b * cC + c)) * cL + l];
    s += X[((size_t)(b * cC + c) * cL + l) * cD + d] * mv;
  }
  float den = fmaxf(msumc[b * cL + l], 1.0f);
  Out[idx] = s / den;
}

// E[b,c,d] = sum_l X[b,c,l,d]*mask[b,c,l] / max(msuml,1)
__global__ void k_mean_l(const float* __restrict__ X, const float* __restrict__ mask,
                         const float* __restrict__ msuml, float* __restrict__ Out) {
  int idx = blockIdx.x * blockDim.x + threadIdx.x;  // B*C*D
  if (idx >= cB * cC * cD) return;
  int d  = idx & (cD - 1);
  int bc = idx >> 7;
  float s = 0.0f;
  const float* xp = X + (size_t)bc * cL * cD + d;
  const float* mp = mask + (size_t)bc * cL;
  for (int l = 0; l < cL; ++l) s += xp[(size_t)l * cD] * mp[l];
  Out[idx] = s / fmaxf(msuml[bc], 1.0f);
}

// Channel scores with mask-overlap scaling.
__global__ void k_chS(const float* __restrict__ q, const float* __restrict__ k,
                      const float* __restrict__ mask, const float* __restrict__ msuml,
                      float* __restrict__ S) {
  int idx = blockIdx.x * blockDim.x + threadIdx.x;  // B*C*C
  if (idx >= cB * cC * cC) return;
  int j = idx & (cC - 1);
  int t = idx >> 6;
  int i = t & (cC - 1);
  int b = t >> 6;
  const float* qp = q + (size_t)(b * cC + i) * cD;
  const float* kp = k + (size_t)(b * cC + j) * cD;
  float s = 0.0f;
  for (int d = 0; d < cD; ++d) s += qp[d] * kp[d];
  s *= 0.08838834764831845f;  // 128^-0.5
  const float* mi = mask + (size_t)(b * cC + i) * cL;
  const float* mj = mask + (size_t)(b * cC + j) * cL;
  float joint = 0.0f;
  for (int l = 0; l < cL; ++l) joint += mi[l] * mj[l];
  float tp = msuml[b * cC + i] + msuml[b * cC + j];
  float ov = 2.0f * joint / fmaxf(tp, 1.0f);
  S[idx] = s * (0.5f + 0.5f * ov);
}

__global__ void k_softmax_rows(float* __restrict__ S, int rows, int n) {
  int r = blockIdx.x * blockDim.x + threadIdx.x;
  if (r >= rows) return;
  float* row = S + (size_t)r * n;
  float mx = -3.402823466e38f;
  for (int j = 0; j < n; ++j) mx = fmaxf(mx, row[j]);
  float sum = 0.0f;
  for (int j = 0; j < n; ++j) {
    float e = expf(row[j] - mx);
    row[j] = e;
    sum += e;
  }
  float inv = 1.0f / sum;
  for (int j = 0; j < n; ++j) row[j] *= inv;
}

// out[b,i,d] = sum_j P[b,i,j] * v[b,j,d]
__global__ void k_attn_apply(const float* __restrict__ P, const float* __restrict__ v,
                             float* __restrict__ Out) {
  int idx = blockIdx.x * blockDim.x + threadIdx.x;  // B*C*D
  if (idx >= cB * cC * cD) return;
  int d  = idx & (cD - 1);
  int bi = idx >> 7;
  int b  = bi >> 6;
  const float* pr = P + (size_t)bi * cC;
  const float* vb = v + (size_t)(b * cC) * cD + d;
  float s = 0.0f;
  for (int j = 0; j < cC; ++j) s += pr[j] * vb[(size_t)j * cD];
  Out[idx] = s;
}

// out[b,c,l,d] = X + T[b,l,d] + Cc[b,c,d]
__global__ void k_bcast_add(const float* __restrict__ X, const float* __restrict__ T,
                            const float* __restrict__ Cc, float* __restrict__ Out, int n) {
  int idx = blockIdx.x * blockDim.x + threadIdx.x;
  if (idx >= n) return;
  int d  = idx & (cD - 1);
  int l  = (idx >> 7) & (cL - 1);
  int bc = idx >> 16;           // b*C + c
  int b  = bc >> 6;
  Out[idx] = X[idx] + T[((size_t)(b * cL + l)) * cD + d] + Cc[(size_t)bc * cD + d];
}

// ---------------------------------------------------------------------------
// Host orchestration
// ---------------------------------------------------------------------------
static inline unsigned gsz(long long n) { return (unsigned)((n + 255) / 256); }

static void gemm(const float* A, const _Float16* Bt, const float* bias, float* Cmat,
                 int M, int N, int K, int act, hipStream_t stream) {
  long long waves = (long long)(M >> 4) * (N >> 6);
  long long threads = waves * 32;
  hipLaunchKernelGGL(k_wmma_gemm, dim3(gsz(threads)), dim3(256), 0, stream,
                     A, Bt, bias, Cmat, M, N, K, act);
}

static void ln_launch(const float* A, const float* Add, const float* g, const float* be,
                      float* Out, int rows, hipStream_t stream) {
  long long threads = (long long)rows * 32;
  hipLaunchKernelGGL(k_ln, dim3(gsz(threads)), dim3(256), 0, stream, A, Add, g, be, Out, rows);
}

extern "C" void kernel_launch(void* const* d_in, const int* in_sizes, int n_in,
                              void* d_out, int out_size, void* d_ws, size_t ws_size,
                              hipStream_t stream) {
  (void)in_sizes; (void)n_in; (void)out_size; (void)ws_size;

  const float* x_in = (const float*)d_in[0];
  const float* mask = (const float*)d_in[1];
  auto P = [&](int i) { return (const float*)d_in[i]; };

  const size_t NF  = (size_t)cB * cC * cL * cD;   // 33,554,432
  const size_t N1f = (size_t)cBC * 256 * cD;
  const size_t N2f = (size_t)cBC * 128 * cD;
  const size_t N3f = (size_t)cBC * 64 * cD;
  const size_t BLD = (size_t)cB * cL * cD;
  const size_t BCD = (size_t)cB * cC * cD;
  const int rows_full = cB * cC * cL;             // 262,144

  // ---- workspace layout (floats) ----
  float* W = (float*)d_ws;
  size_t off = 0;
  auto alloc = [&](size_t nf) { float* p = W + off; off += nf; return p; };
  float* xcur = alloc(NF);
  float* ybuf = alloc(NF);
  float* zbuf = alloc(NF);
  float* s0   = alloc(NF);
  float* t0   = alloc(NF);
  float* hid  = alloc(4 * NF);   // FFN hidden; multiscale pyramid overlaid here
  // pyramid buffers live only before ms_ffn uses `hid` -> safe overlay
  float* s1 = hid;
  float* t1 = s1 + N1f;
  float* s2 = t1 + N1f;
  float* t2 = s2 + N2f;
  float* s3 = t2 + N2f;
  float* t3 = s3 + N3f;
  float* Et_raw = alloc(BLD);
  float* Et     = alloc(BLD);
  float* cE  = alloc(BCD);
  float* cq  = alloc(BCD);
  float* ck  = alloc(BCD);
  float* cv  = alloc(BCD);
  float* cA  = alloc(BCD);
  float* cO  = alloc(BCD);
  float* cE1 = alloc(BCD);
  float* cF  = alloc(BCD);
  float* Ec  = alloc(BCD);
  float* cHid  = alloc((size_t)cBC * 512);
  float* Smat  = alloc((size_t)cB * cC * cC);
  float* msumc = alloc((size_t)cB * cL);
  float* msuml = alloc((size_t)cB * cC);
  float* Tlin  = alloc(BLD);
  float* Clin  = alloc(BCD);

  _Float16* wh = (_Float16*)(W + off);
  size_t hoff = 0;
  auto halloc = [&](size_t n) { _Float16* p = wh + hoff; hoff += n; return p; };
  _Float16* lcpw16   = halloc(cD * cD);
  _Float16* sdpw16[3], *supw16[3], *tupw16[3];
  for (int i = 0; i < 3; ++i) { sdpw16[i] = halloc(cD * cD); supw16[i] = halloc(cD * cD); tupw16[i] = halloc(cD * cD); }
  _Float16* msw1_16 = halloc((size_t)cD * 512);
  _Float16* msw2_16 = halloc((size_t)512 * cD);
  _Float16* thw16[4]; for (int i = 0; i < 4; ++i) thw16[i] = halloc(cD * cD);
  _Float16* chw16[4]; for (int i = 0; i < 4; ++i) chw16[i] = halloc(cD * cD);
  _Float16* chf1_16 = halloc((size_t)cD * 512);
  _Float16* chf2_16 = halloc((size_t)512 * cD);
  _Float16* hnt16 = halloc(cD * cD);
  _Float16* hnc16 = halloc(cD * cD);
  _Float16* hnf1_16 = halloc((size_t)cD * 512);
  _Float16* hnf2_16 = halloc((size_t)512 * cD);

  // ---- prep all weights into f16 [N,K] ----
  auto prep = [&](int srcIdx, _Float16* dst, int Kd, int Nd, int tr) {
    int n = Kd * Nd;
    hipLaunchKernelGGL(k_prep_w, dim3(gsz(n)), dim3(256), 0, stream, P(srcIdx), dst, Kd, Nd, tr);
  };
  prep(4, lcpw16, cD, cD, 0);                       // lc_pw_w [O,C] already [N,K]
  const int sdIdx[3] = {8, 20, 32};                 // sd{i} base (dw_w)
  for (int i = 0; i < 3; ++i) {
    prep(sdIdx[i] + 2,  sdpw16[i], cD, cD, 0);      // sd{i}_pw_w
    prep(sdIdx[i] + 6,  supw16[i], cD, cD, 0);      // su{i}_pw_w
    prep(sdIdx[i] + 10, tupw16[i], cD, cD, 0);      // tu{i}_pw_w
  }
  prep(44, msw1_16, cD, 512, 1);                    // ms_ffn_w1 [D,4D]
  prep(46, msw2_16, 512, cD, 1);                    // ms_ffn_w2 [4D,D]
  for (int i = 0; i < 4; ++i) prep(50 + 2 * i, thw16[i], cD, cD, 1);  // th q,k,v,o
  for (int i = 0; i < 4; ++i) prep(60 + 2 * i, chw16[i], cD, cD, 1);  // ch q,k,v,o
  prep(72, chf1_16, cD, 512, 1);
  prep(74, chf2_16, 512, cD, 1);
  prep(76, hnt16, cD, cD, 1);
  prep(77, hnc16, cD, cD, 1);
  prep(80, hnf1_16, cD, 512, 1);
  prep(82, hnf2_16, 512, cD, 1);

  // ================= stage 1: local conv =================
  hipLaunchKernelGGL(k_dwconv_gelu, dim3(gsz(NF)), dim3(256), 0, stream,
                     x_in, P(2), P(3), ybuf, cBC, cL, cKC);
  gemm(ybuf, lcpw16, P(5), zbuf, rows_full, cD, cD, 0, stream);
  ln_launch(zbuf, nullptr, P(6), P(7), ybuf, rows_full, stream);
  hipLaunchKernelGGL(k_add, dim3(gsz(NF)), dim3(256), 0, stream, x_in, ybuf, xcur, (int)NF);

  // ================= stage 2: multiscale =================
  hipLaunchKernelGGL(k_decomp, dim3(gsz(NF)), dim3(256), 0, stream, xcur, s0, t0, cBC, 512);
  hipLaunchKernelGGL(k_avgpool2, dim3(gsz(N1f)), dim3(256), 0, stream, xcur, ybuf, cBC, 512);
  hipLaunchKernelGGL(k_decomp, dim3(gsz(N1f)), dim3(256), 0, stream, ybuf, s1, t1, cBC, 256);
  hipLaunchKernelGGL(k_avgpool2, dim3(gsz(N2f)), dim3(256), 0, stream, ybuf, zbuf, cBC, 256);
  hipLaunchKernelGGL(k_decomp, dim3(gsz(N2f)), dim3(256), 0, stream, zbuf, s2, t2, cBC, 128);
  hipLaunchKernelGGL(k_avgpool2, dim3(gsz(N3f)), dim3(256), 0, stream, zbuf, ybuf, cBC, 128);
  hipLaunchKernelGGL(k_decomp, dim3(gsz(N3f)), dim3(256), 0, stream, ybuf, s3, t3, cBC, 64);

  float* seas[4] = {s0, s1, s2, s3};
  float* tres[4] = {t0, t1, t2, t3};
  const int Ls[4] = {512, 256, 128, 64};

  // downward season mixing
  for (int i = 0; i < 3; ++i) {
    int Lo = Ls[i + 1];
    long long n = (long long)cBC * Lo * cD;
    hipLaunchKernelGGL(k_avgpool2, dim3(gsz(n)), dim3(256), 0, stream, seas[i], ybuf, cBC, Ls[i]);
    hipLaunchKernelGGL(k_dwconv_gelu, dim3(gsz(n)), dim3(256), 0, stream,
                       ybuf, P(sdIdx[i] + 0), P(sdIdx[i] + 1), zbuf, cBC, Lo, 3);
    gemm(zbuf, sdpw16[i], P(sdIdx[i] + 3), ybuf, cBC * Lo, cD, cD, 0, stream);
    hipLaunchKernelGGL(k_add, dim3(gsz(n)), dim3(256), 0, stream, seas[i + 1], ybuf, seas[i + 1], (int)n);
  }
  // upward season mixing
  for (int i = 2; i >= 0; --i) {
    int Lo = Ls[i];
    long long n = (long long)cBC * Lo * cD;
    hipLaunchKernelGGL(k_interp2x, dim3(gsz(n)), dim3(256), 0, stream, seas[i + 1], ybuf, cBC, Ls[i + 1]);
    hipLaunchKernelGGL(k_dwconv_gelu, dim3(gsz(n)), dim3(256), 0, stream,
                       ybuf, P(sdIdx[i] + 4), P(sdIdx[i] + 5), zbuf, cBC, Lo, 3);
    gemm(zbuf, supw16[i], P(sdIdx[i] + 7), ybuf, cBC * Lo, cD, cD, 0, stream);
    hipLaunchKernelGGL(k_add, dim3(gsz(n)), dim3(256), 0, stream, seas[i], ybuf, seas[i], (int)n);
  }
  // upward trend mixing
  for (int i = 2; i >= 0; --i) {
    int Lo = Ls[i];
    long long n = (long long)cBC * Lo * cD;
    hipLaunchKernelGGL(k_interp2x, dim3(gsz(n)), dim3(256), 0, stream, tres[i + 1], ybuf, cBC, Ls[i + 1]);
    hipLaunchKernelGGL(k_dwconv_gelu, dim3(gsz(n)), dim3(256), 0, stream,
                       ybuf, P(sdIdx[i] + 8), P(sdIdx[i] + 9), zbuf, cBC, Lo, 3);
    gemm(zbuf, tupw16[i], P(sdIdx[i] + 11), ybuf, cBC * Lo, cD, cD, 0, stream);
    hipLaunchKernelGGL(k_add, dim3(gsz(n)), dim3(256), 0, stream, tres[i], ybuf, tres[i], (int)n);
  }
  // out = s0 + t0; h += ffn(LN(out))   (pyramid region now free -> hid reuse)
  hipLaunchKernelGGL(k_add, dim3(gsz(NF)), dim3(256), 0, stream, s0, t0, ybuf, (int)NF);
  ln_launch(ybuf, nullptr, P(48), P(49), zbuf, rows_full, stream);
  gemm(zbuf, msw1_16, P(45), hid, rows_full, 512, cD, 1, stream);   // GELU fused
  gemm(hid, msw2_16, P(47), zbuf, rows_full, cD, 512, 0, stream);
  hipLaunchKernelGGL(k_add, dim3(gsz(NF)), dim3(256), 0, stream, xcur, zbuf, xcur, (int)NF);

  // ================= stage 3: temporal HE =================
  gemm(xcur, thw16[0], P(51), s0, rows_full, cD, cD, 0, stream);    // Q
  gemm(xcur, thw16[1], P(53), t0, rows_full, cD, cD, 0, stream);    // K
  gemm(xcur, thw16[2], P(55), ybuf, rows_full, cD, cD, 0, stream);  // V
  hipLaunchKernelGGL(k_tattn, dim3(gsz((long long)cB * cL * cH * cC)), dim3(256), 0, stream,
                     s0, t0, ybuf, mask, zbuf);
  gemm(zbuf, thw16[3], P(57), s0, rows_full, cD, cD, 0, stream);    // O proj
  hipLaunchKernelGGL(k_msum_c, dim3(gsz(cB * cL)), dim3(256), 0, stream, mask, msumc);
  hipLaunchKernelGGL(k_mean_c, dim3(gsz((long long)BLD)), dim3(256), 0, stream, s0, mask, msumc, Et_raw);
  ln_launch(Et_raw, nullptr, P(58), P(59), Et, cB * cL, stream);

  // ================= stage 4: channel HE =================
  hipLaunchKernelGGL(k_msum_l, dim3(gsz(cB * cC)), dim3(256), 0, stream, mask, msuml);
  hipLaunchKernelGGL(k_mean_l, dim3(gsz((long long)BCD)), dim3(256), 0, stream, xcur, mask, msuml, cE);
  gemm(cE, chw16[0], P(61), cq, cB * cC, cD, cD, 0, stream);
  gemm(cE, chw16[1], P(63), ck, cB * cC, cD, cD, 0, stream);
  gemm(cE, chw16[2], P(65), cv, cB * cC, cD, cD, 0, stream);
  hipLaunchKernelGGL(k_chS, dim3(gsz((long long)cB * cC * cC)), dim3(256), 0, stream,
                     cq, ck, mask, msuml, Smat);
  hipLaunchKernelGGL(k_softmax_rows, dim3(gsz(cB * cC)), dim3(256), 0, stream, Smat, cB * cC, cC);
  hipLaunchKernelGGL(k_attn_apply, dim3(gsz((long long)BCD)), dim3(256), 0, stream, Smat, cv, cA);
  gemm(cA, chw16[3], P(67), cO, cB * cC, cD, cD, 0, stream);
  ln_launch(cE, cO, P(68), P(69), cE1, cB * cC, stream);            // n1: LN(E + out)
  gemm(cE1, chf1_16, P(73), cHid, cB * cC, 512, cD, 1, stream);
  gemm(cHid, chf2_16, P(75), cF, cB * cC, cD, 512, 0, stream);
  ln_launch(cE1, cF, P(70), P(71), Ec, cB * cC, stream);            // n2: LN(E1 + ffn)

  // ================= stage 5: HE -> node =================
  gemm(Et, hnt16, nullptr, Tlin, cB * cL, cD, cD, 0, stream);
  gemm(Ec, hnc16, nullptr, Clin, cB * cC, cD, cD, 0, stream);
  hipLaunchKernelGGL(k_bcast_add, dim3(gsz(NF)), dim3(256), 0, stream, xcur, Tlin, Clin, ybuf, (int)NF);
  ln_launch(ybuf, nullptr, P(78), P(79), zbuf, rows_full, stream);
  gemm(zbuf, hnf1_16, P(81), hid, rows_full, 512, cD, 1, stream);
  gemm(hid, hnf2_16, P(83), zbuf, rows_full, cD, 512, 0, stream);
  hipLaunchKernelGGL(k_add, dim3(gsz(NF)), dim3(256), 0, stream, ybuf, zbuf, (float*)d_out, (int)NF);
}
